// AttentionOperation_28114855920200
// MI455X (gfx1250) — compile-verified, matrix operand order checked by reading, not run on hardware
//
#include <hip/hip_runtime.h>
#include <hip/hip_bf16.h>
#include <math.h>

// ---------------------------------------------------------------------------
// Fused LN + multi-head attention block for MI455X (gfx1250, wave32, WMMA).
// GEMMs: v_wmma_f32_16x16x32_f16, 16x64 tile per wave (A-fragment reuse x4).
// Attention: flash-style; K tile staged via Tensor Data Mover
// (tensor_load_to_lds + s_wait_tensorcnt), V staged transposed via LDS.
// ---------------------------------------------------------------------------

#define BB 4
#define CC 512
#define TT 2048
#define HH 8
#define DD 64          // head dim
#define MROWS (BB*TT)  // 8192

typedef __attribute__((ext_vector_type(16))) _Float16 v16h;
typedef __attribute__((ext_vector_type(8)))  _Float16 v8h;
typedef __attribute__((ext_vector_type(8)))  float    v8f;
typedef __attribute__((ext_vector_type(4)))  unsigned int u32x4;
typedef __attribute__((ext_vector_type(8)))  int      i32x8;
typedef __attribute__((ext_vector_type(4)))  int      i32x4;

__device__ __forceinline__ v8f wmma_f16(v16h a, v16h b, v8f c) {
    // 8 args: (neg_a, A, neg_b, B, c_mod, C, reuse_a, reuse_b)
    return __builtin_amdgcn_wmma_f32_16x16x32_f16(false, a, false, b,
                                                  (short)0, c, false, false);
}

// Load a 16x32 f16 fragment (A-matrix layout, also valid as B when the source
// S has the K-dimension contiguous per output column).
// ISA layout: lane<16 holds row (lane&15), K = {0..7, 16..23};
//             lane>=16 holds row (lane&15), K = {8..15, 24..31}.
__device__ __forceinline__ v16h load_frag(const _Float16* S, int row_base,
                                          int row_stride, int k0) {
    int lane = threadIdx.x & 31;
    int r    = row_base + (lane & 15);
    int koff = (lane >> 4) * 8;
    const _Float16* p = S + (size_t)r * row_stride + k0 + koff;
    v8h lo = *(const v8h*)(p);        // K = k0+koff .. +7      (16B aligned)
    v8h hi = *(const v8h*)(p + 16);   // K = k0+16+koff .. +7   (16B aligned)
    v16h a;
#pragma unroll
    for (int i = 0; i < 8; ++i) { a[i] = lo[i]; a[8 + i] = hi[i]; }
    return a;
}

// ---------------------------------------------------------------------------
// Kernel 1: convert the four 512x512 fp32 weight matrices to f16 (q,k,v,o).
// ---------------------------------------------------------------------------
__global__ void k_convert_w(const float* __restrict__ Wq,
                            const float* __restrict__ Wk,
                            const float* __restrict__ Wv,
                            const float* __restrict__ Wo,
                            _Float16* __restrict__ wh) {
    int idx = blockIdx.x * blockDim.x + threadIdx.x;   // < 4*512*512
    int sel = idx >> 18;                               // C*C = 2^18
    int off = idx & ((CC * CC) - 1);
    const float* src = (sel == 0) ? Wq : (sel == 1) ? Wk : (sel == 2) ? Wv : Wo;
    wh[idx] = (_Float16)src[off];
}

// ---------------------------------------------------------------------------
// Kernel 2: transpose + LayerNorm.  x[B,C,T] fp32 -> xh[B*T, C] f16.
// One thread per t; reads over C are coalesced across the t-dimension.
// ---------------------------------------------------------------------------
__global__ void k_layernorm(const float* __restrict__ x,
                            const float* __restrict__ gamma,
                            const float* __restrict__ beta,
                            _Float16* __restrict__ xh) {
    int b = blockIdx.y;
    int t = blockIdx.x * blockDim.x + threadIdx.x;
    const float* xb = x + (size_t)b * CC * TT;

    float s = 0.f, s2 = 0.f;
#pragma unroll 4
    for (int c = 0; c < CC; ++c) {
        float v = xb[(size_t)c * TT + t];
        s += v; s2 += v * v;
    }
    float mu   = s * (1.0f / CC);
    float var  = s2 * (1.0f / CC) - mu * mu;
    float rsig = rsqrtf(var + 1e-5f);

    _Float16* dst = xh + ((size_t)b * TT + t) * CC;
#pragma unroll 4
    for (int c = 0; c < CC; ++c) {
        float v = xb[(size_t)c * TT + t];
        dst[c] = (_Float16)((v - mu) * rsig * gamma[c] + beta[c]);
    }
}

// ---------------------------------------------------------------------------
// Kernel 3: QKV projection.  y = xn @ W^T + b.
// grid = (M/128, C/64, 3).  Block = 256 = 8 waves; each wave owns a 16x64
// output tile: per K-step 1 A-frag + 4 B-frags + 4 WMMAs (A reused 4x).
// Output written as f16 [B,H,T,D] for the attention kernel.
// ---------------------------------------------------------------------------
__global__ void k_qkv_gemm(const _Float16* __restrict__ xh,
                           const _Float16* __restrict__ wh,   // 4 matrices
                           const float* __restrict__ bq,
                           const float* __restrict__ bk,
                           const float* __restrict__ bv,
                           _Float16* __restrict__ qkvh) {
    int w    = threadIdx.x >> 5;
    int lane = threadIdx.x & 31;
    int z    = blockIdx.z;                       // 0=q 1=k 2=v
    int m0   = (blockIdx.x * 8 + w) * 16;        // row tile in [0, B*T)
    int n0   = blockIdx.y * 64;                  // col quad-tile in [0, C)

    const _Float16* W = wh + (size_t)z * CC * CC;
    const float* bias = (z == 0) ? bq : (z == 1) ? bk : bv;

    v8f acc[4] = {};
#pragma unroll 2
    for (int kt = 0; kt < CC / 32; ++kt) {
        v16h a = load_frag(xh, m0, CC, kt * 32);
#pragma unroll
        for (int j = 0; j < 4; ++j)
            acc[j] = wmma_f16(a, load_frag(W, n0 + j * 16, CC, kt * 32), acc[j]);
    }

    int col = lane & 15, hi = lane >> 4;
    _Float16* dst = qkvh + (size_t)z * MROWS * CC;
#pragma unroll
    for (int j = 0; j < 4; ++j) {
        int n = n0 + j * 16 + col;
        int h = n >> 6, d = n & 63;
        float bn = bias[n];
#pragma unroll
        for (int r = 0; r < 8; ++r) {
            int m = m0 + r + 8 * hi;
            int bidx = m >> 11;          // /T
            int t    = m & (TT - 1);
            dst[(((size_t)(bidx * HH + h)) * TT + t) * DD + d] =
                (_Float16)(acc[j][r] + bn);
        }
    }
}

// ---------------------------------------------------------------------------
// Kernel 4: flash attention.  grid = (T/128, B*H), block = 256 (8 waves).
// Wave w handles query rows [q0, q0+16).  K tile (32x64 f16, row stride 64)
// staged via the Tensor Data Mover; V staged transposed via vector LDS ops.
// ---------------------------------------------------------------------------
__global__ void k_flash_attn(const _Float16* __restrict__ qkvh,
                             _Float16* __restrict__ oh) {
    __shared__ _Float16 Kt[32][DD];        // [key_local][d]        4 KB (TDM dest)
    __shared__ _Float16 Vt[DD][32];        // [d][key_local]        4 KB
    __shared__ _Float16 Pb[8][16][32];     // per-wave P tile       8 KB

    int tid  = threadIdx.x;
    int w    = tid >> 5;
    int lane = tid & 31;
    int col  = lane & 15, hi = lane >> 4;

    int bh = blockIdx.y;                   // b*H + h
    int b  = bh >> 3, h = bh & 7;
    size_t base = (size_t)bh * TT * DD;
    const _Float16* Qh = qkvh + 0 * (size_t)MROWS * CC + base;
    const _Float16* Kh = qkvh + 1 * (size_t)MROWS * CC + base;
    const _Float16* Vh = qkvh + 2 * (size_t)MROWS * CC + base;

    int q0 = blockIdx.x * 128 + w * 16;    // query-row tile of this wave

    // Q fragments for d = 0..31 and 32..63 (loaded once, reused T/32 times)
    v16h aq0 = load_frag(Qh, q0, DD, 0);
    v16h aq1 = load_frag(Qh, q0, DD, 32);

    v8f acc[4] = {};                       // 16 x 64 output accumulator
    float mrun[8], lrun[8];
#pragma unroll
    for (int r = 0; r < 8; ++r) { mrun[r] = -INFINITY; lrun[r] = 0.f; }

    const float scale = 0.125f;            // 1/sqrt(64)
    // cooperative stage indices for V: 256 threads x 8 halves = 32x64 tile
    int kr = (tid * 8) >> 6;
    int kc = (tid * 8) & 63;

    // Static part of the TDM descriptor for the K tile (uniform values).
    unsigned int lds_k = (unsigned int)(uintptr_t)&Kt[0][0]; // LDS byte offset
    i32x8 g1;
    g1[0] = (1 << 16);                     // workgroup_mask=0, data_size=1 (2B)
    g1[1] = (DD << 16);                    // tensor_dim0[15:0]=64; bar addr=0
    g1[2] = (TT << 16) | (DD >> 16);       // tensor_dim0 hi | tensor_dim1 lo
    g1[3] = (DD << 16) | (TT >> 16);       // tensor_dim1 hi | tile_dim0=64
    g1[4] = 32;                            // tile_dim1=32, tile_dim2=0
    g1[5] = DD;                            // tensor_dim0_stride = 64 elements
    g1[6] = 0;                             // stride0 hi | stride1 lo
    g1[7] = 0;                             // stride1 hi
    i32x4 gz = {0, 0, 0, 0};               // groups 2/3 unused (2D tensor)
    i32x8 gz8 = {0, 0, 0, 0, 0, 0, 0, 0};  // extra group (6-arg toolchain form)

    for (int kb = 0; kb < TT / 32; ++kb) {
        __syncthreads();                   // previous iter done reading LDS

        // ---- K tile via Tensor Data Mover (wave 0 issues, TENSORcnt) ----
        if (w == 0) {
            unsigned long long ga =
                (unsigned long long)(uintptr_t)(Kh + (size_t)(kb * 32) * DD);
            u32x4 g0;
            g0[0] = 1u;                                    // count=1, no gather
            g0[1] = lds_k;                                 // lds_addr
            g0[2] = (unsigned int)(ga & 0xFFFFFFFFu);      // global_addr lo
            g0[3] = (unsigned int)((ga >> 32) & 0x01FFFFFFu) | (2u << 30); // type=2
            __builtin_amdgcn_tensor_load_to_lds(g0, g1, gz, gz, gz8, 0);
        }

        // ---- V tile staged transposed by all 256 threads ----
        const _Float16* gv = Vh + ((size_t)(kb * 32 + kr)) * DD + kc;
        v8h vv = *(const v8h*)gv;
#pragma unroll
        for (int i = 0; i < 8; ++i) Vt[kc + i][kr] = vv[i];   // transpose V
        if (kb + 1 < TT / 32)
            __builtin_prefetch(gv + 32 * DD, 0, 1);           // global_prefetch_b8

        if (w == 0) __builtin_amdgcn_s_wait_tensorcnt(0);     // TDM complete
        __syncthreads();                                      // publish K + V

        // ---- scores: S(16x32) = Q(16x64) @ K^T, two 16x16 n-tiles ----
        v8f z = {};
        v8f s0 = wmma_f16(aq0, load_frag(&Kt[0][0],  0, DD, 0),  z);
        s0     = wmma_f16(aq1, load_frag(&Kt[0][0],  0, DD, 32), s0);
        v8f s1 = wmma_f16(aq0, load_frag(&Kt[0][0], 16, DD, 0),  z);
        s1     = wmma_f16(aq1, load_frag(&Kt[0][0], 16, DD, 32), s1);

        // ---- online softmax (rows live in regs r, spread over 16 lanes) ----
        float p0[8], p1[8], corr[8];
#pragma unroll
        for (int r = 0; r < 8; ++r) {
            float a = s0[r] * scale, c = s1[r] * scale;
            float rm = fmaxf(a, c);
#pragma unroll
            for (int msk = 1; msk <= 8; msk <<= 1)
                rm = fmaxf(rm, __shfl_xor(rm, msk, 32));
            float mnew = fmaxf(mrun[r], rm);
            corr[r] = __expf(mrun[r] - mnew);
            p0[r]   = __expf(a - mnew);
            p1[r]   = __expf(c - mnew);
            float rs = p0[r] + p1[r];
#pragma unroll
            for (int msk = 1; msk <= 8; msk <<= 1)
                rs += __shfl_xor(rs, msk, 32);
            lrun[r] = lrun[r] * corr[r] + rs;
            mrun[r] = mnew;
        }
#pragma unroll
        for (int v = 0; v < 4; ++v)
#pragma unroll
            for (int r = 0; r < 8; ++r) acc[v][r] *= corr[r];

        // ---- D-frag -> A-frag relayout of P through per-wave LDS ----
        _Float16* Pw = &Pb[w][0][0];
#pragma unroll
        for (int r = 0; r < 8; ++r) {
            int row = r + 8 * hi;
            Pw[row * 32 + col]      = (_Float16)p0[r];
            Pw[row * 32 + col + 16] = (_Float16)p1[r];
        }
        v16h pa = load_frag(Pw, 0, 32, 0);   // compiler inserts ds waits

        // ---- acc += P(16x32) @ V(32x64), four 16-col tiles ----
#pragma unroll
        for (int v = 0; v < 4; ++v)
            acc[v] = wmma_f16(pa, load_frag(&Vt[0][0], v * 16, 32, 0), acc[v]);
    }

    // normalize and store O as f16 [B*T, C]
    float inv[8];
#pragma unroll
    for (int r = 0; r < 8; ++r) inv[r] = 1.0f / lrun[r];
#pragma unroll
    for (int v = 0; v < 4; ++v)
#pragma unroll
        for (int r = 0; r < 8; ++r) {
            int t = q0 + r + 8 * hi;
            int c = h * DD + v * 16 + col;
            oh[((size_t)b * TT + t) * CC + c] = (_Float16)(acc[v][r] * inv[r]);
        }
}

// ---------------------------------------------------------------------------
// Kernel 5: output projection + residual + transpose back to [B,C,T] fp32.
// 16x64 tile per wave, A-fragment reuse x4.
// ---------------------------------------------------------------------------
__global__ void k_out_proj(const _Float16* __restrict__ oh,
                           const _Float16* __restrict__ wh,   // Wo at slot 3
                           const float* __restrict__ bo,
                           const float* __restrict__ x,
                           float* __restrict__ out) {
    int w    = threadIdx.x >> 5;
    int lane = threadIdx.x & 31;
    int m0   = (blockIdx.x * 8 + w) * 16;
    int n0   = blockIdx.y * 64;

    const _Float16* Wo = wh + (size_t)3 * CC * CC;
    v8f acc[4] = {};
#pragma unroll 2
    for (int kt = 0; kt < CC / 32; ++kt) {
        v16h a = load_frag(oh, m0, CC, kt * 32);
#pragma unroll
        for (int j = 0; j < 4; ++j)
            acc[j] = wmma_f16(a, load_frag(Wo, n0 + j * 16, CC, kt * 32), acc[j]);
    }

    int col = lane & 15, hi = lane >> 4;
#pragma unroll
    for (int j = 0; j < 4; ++j) {
        int c = n0 + j * 16 + col;
        float bn = bo[c];
#pragma unroll
        for (int r = 0; r < 8; ++r) {
            int m = m0 + r + 8 * hi;
            int bidx = m >> 11;
            int t    = m & (TT - 1);
            size_t idx = ((size_t)bidx * CC + c) * TT + t;
            out[idx] = acc[j][r] + bn + x[idx];
        }
    }
}

// ---------------------------------------------------------------------------
extern "C" void kernel_launch(void* const* d_in, const int* in_sizes, int n_in,
                              void* d_out, int out_size, void* d_ws, size_t ws_size,
                              hipStream_t stream) {
    const float* x  = (const float*)d_in[0];
    const float* Wq = (const float*)d_in[1];
    const float* bq = (const float*)d_in[2];
    const float* Wk = (const float*)d_in[3];
    const float* bk = (const float*)d_in[4];
    const float* Wv = (const float*)d_in[5];
    const float* bv = (const float*)d_in[6];
    const float* Wo = (const float*)d_in[7];
    const float* bo = (const float*)d_in[8];
    const float* g  = (const float*)d_in[9];
    const float* be = (const float*)d_in[10];
    float* out = (float*)d_out;

    // Workspace layout (bytes): xh 8MB | wh 2MB | qkv 24MB | oh 8MB  (~42 MB)
    char* ws = (char*)d_ws;
    _Float16* xh   = (_Float16*)(ws);
    _Float16* wh   = (_Float16*)(ws + (size_t)MROWS * CC * 2);
    _Float16* qkvh = (_Float16*)(ws + (size_t)MROWS * CC * 2 + (size_t)4 * CC * CC * 2);
    _Float16* oh   = (_Float16*)(ws + (size_t)MROWS * CC * 2 + (size_t)4 * CC * CC * 2
                                    + (size_t)3 * MROWS * CC * 2);

    // 1) weights -> f16
    k_convert_w<<<dim3((4 * CC * CC) / 256), dim3(256), 0, stream>>>(Wq, Wk, Wv, Wo, wh);
    // 2) transpose + layernorm
    k_layernorm<<<dim3(TT / 256, BB), dim3(256), 0, stream>>>(x, g, be, xh);
    // 3) QKV projections (z selects matrix)
    k_qkv_gemm<<<dim3(MROWS / 128, CC / 64, 3), dim3(256), 0, stream>>>(
        xh, wh, bq, bk, bv, qkvh);
    // 4) flash attention (TDM-staged K tiles)
    k_flash_attn<<<dim3(TT / 128, BB * HH), dim3(256), 0, stream>>>(qkvh, oh);
    // 5) output projection + residual + transpose
    k_out_proj<<<dim3(MROWS / 128, CC / 64), dim3(256), 0, stream>>>(
        oh, wh, bo, x, out);
}
